// LanguageModel_54400055771802
// MI455X (gfx1250) — matrix operand order repaired and checked by reading, not compiled, not run on hardware
//
#include <hip/hip_runtime.h>
#include <stdint.h>

// ---------------------------------------------------------------------------
// Model constants (match reference)
// ---------------------------------------------------------------------------
static constexpr long long cV  = 32000;
static constexpr long long cD  = 1024;
static constexpr long long cT  = 2048;
static constexpr long long cH  = 4;
static constexpr long long cDH = 256;
static constexpr long long cL  = 3;
static constexpr long long cB  = 2;
static constexpr long long cFF = 4096;
static constexpr long long cBT = cB * cT;   // 4096 rows
static constexpr long long cBH = cB * cH;   // 8 attention batches

// ---------------------------------------------------------------------------
// Types for WMMA (gfx1250, wave32)
// ---------------------------------------------------------------------------
typedef __bf16          v16bf __attribute__((ext_vector_type(16)));
typedef float           v8f   __attribute__((ext_vector_type(8)));
typedef uint32_t        u32x4 __attribute__((ext_vector_type(4)));
typedef unsigned short  u16x8 __attribute__((ext_vector_type(8)));

__device__ __host__ __forceinline__ unsigned short f32_to_bf16(float f) {
    union { float f; uint32_t u; } x; x.f = f;
    uint32_t u = x.u;
    uint32_t r = u + 0x7FFFu + ((u >> 16) & 1u);   // round-to-nearest-even
    return (unsigned short)(r >> 16);
}

// ---- CDNA5 async global->LDS copy (ASYNCcnt), 16B per lane -----------------
__device__ __forceinline__ void async_copy16(uint32_t lds_addr, const void* gptr) {
    asm volatile("global_load_async_to_lds_b128 %0, %1, off"
                 :: "v"(lds_addr), "v"(gptr) : "memory");
}
__device__ __forceinline__ void wait_async0() {
    asm volatile("s_wait_asynccnt 0" ::: "memory");
}

// ---------------------------------------------------------------------------
// Batched fp32 [K,N] -> bf16 [N,K] transpose-convert (weights become K-major
// so GEMM B-tiles can be staged with contiguous async copies).
// grid = (N/32, K/32, nmat), block = (32,8)
// ---------------------------------------------------------------------------
__global__ void cvtT_f32_bf16(const float* __restrict__ in,
                              unsigned short* __restrict__ out,
                              int K, int N) {
    __shared__ float tile[32][33];
    const long long mat = blockIdx.z;
    const float* src = in + mat * (long long)K * N;
    unsigned short* dst = out + mat * (long long)K * N;
    const int n0 = blockIdx.x * 32, k0 = blockIdx.y * 32;
#pragma unroll
    for (int r = 0; r < 32; r += 8)
        tile[threadIdx.y + r][threadIdx.x] =
            src[(long long)(k0 + threadIdx.y + r) * N + n0 + threadIdx.x];
    __syncthreads();
#pragma unroll
    for (int r = 0; r < 32; r += 8)
        dst[(long long)(n0 + threadIdx.y + r) * K + k0 + threadIdx.x] =
            f32_to_bf16(tile[threadIdx.x][threadIdx.y + r]);
}

// ---------------------------------------------------------------------------
// Embedding: x = tok_emb[idx] + pos_emb ; also produce bf16 copy
// ---------------------------------------------------------------------------
__global__ void embed_kernel(const int* __restrict__ idx,
                             const float* __restrict__ tok,
                             const float* __restrict__ pos,
                             float* __restrict__ xf,
                             unsigned short* __restrict__ xh) {
    long long i = (long long)blockIdx.x * blockDim.x + threadIdx.x; // over BT*D
    long long d  = i % cD;
    long long bt = i / cD;
    long long t  = bt % cT;
    int token = idx[bt];
    float v = tok[(long long)token * cD + d] + pos[t * cD + d];
    xf[i] = v;
    xh[i] = f32_to_bf16(v);
}

// ---------------------------------------------------------------------------
// Generic bf16 WMMA GEMM:  C[M,N](f32 acc) = A[M,K] * Bt[N,K]^T
// (B always stored K-major).  Async double-buffered LDS staging, software
// pipelined (peeled) so the steady-state loop is branch-free.
// z-batched; per-operand base = (z%zmod)*Lo + (z/zmod)*Hi.
// All M,N multiples of 128 and K multiples of 64 in this model -> no guards.
// ---------------------------------------------------------------------------
struct GemmParams {
    const unsigned short* A;  long long lda, sAzLo, sAzHi;
    const unsigned short* Bm; long long ldb, sBzLo, sBzHi;
    float* Cf; unsigned short* Ch; long long ldc;
    int storeCT; long long ldct;      // if set: Ch[col*ldct + row] (transposed store)
    int zmod; long long sCzLo, sCzHi;
    const float* bias;                // per-column, may be null
    const float* resid;               // fp32 residual, may be null
    long long ldr;
    int relu;
    int M, N, K;
};

static constexpr int GBM = 128, GBN = 128, GBK = 64;
static constexpr int LDSS = GBK + 8;          // padded row stride (ushorts); 144B rows keep 16B align
static constexpr int TILE_USH = GBM * LDSS;   // ushorts per buffer

// A-fragment (16x32 bf16) per ISA 7.12.2: lanes 0-15 -> K {s..s+7, s+16..s+23},
// lanes 16-31 -> K {s+8..s+15, s+24..s+31}, row M = lane%16.
__device__ __forceinline__ v16bf load_fragA(const unsigned short* rowp, int s, int lane) {
    const int koff = s + ((lane & 16) ? 8 : 0);
    u32x4 lo = *reinterpret_cast<const u32x4*>(rowp + koff);
    u32x4 hi = *reinterpret_cast<const u32x4*>(rowp + koff + 16);
    struct Pair { u32x4 a, b; } pr{lo, hi};
    return __builtin_bit_cast(v16bf, pr);
}

// B-fragment (32x16 bf16): lanes 0-15 -> K {s..s+7, s+16..s+23}, lanes 16-31 -> K {s+8..s+15, s+24..s+31},
// column N = lane%16 (B-matrix striping per ISA 7.12.4/7.12.5 pattern).
__device__ __forceinline__ v16bf load_fragB(const unsigned short* rowp, int s, int lane) {
    const int koff = s + ((lane & 16) ? 8 : 0);
    u32x4 lo = *reinterpret_cast<const u32x4*>(rowp + koff);
    u32x4 hi = *reinterpret_cast<const u32x4*>(rowp + koff + 16);
    struct Pair { u32x4 a, b; } pr{lo, hi};
    return __builtin_bit_cast(v16bf, pr);
}

__global__ __launch_bounds__(256) void gemm_bf16_kernel(GemmParams p) {
    __shared__ __align__(16) unsigned short As[2 * TILE_USH];
    __shared__ __align__(16) unsigned short Bs[2 * TILE_USH];

    const int tid  = threadIdx.x;
    const int lane = tid & 31;
    const int wave = tid >> 5;
    const int wm   = (wave & 3) * 32;   // 4 wave-rows x 32 rows
    const int wn   = (wave >> 2) * 64;  // 2 wave-cols x 64 cols
    const int z    = blockIdx.z;
    const long long zl = z % p.zmod, zh = z / p.zmod;
    const long long m0 = (long long)blockIdx.y * GBM;
    const long long n0 = (long long)blockIdx.x * GBN;

    const unsigned short* Ag = p.A  + zl * p.sAzLo + zh * p.sAzHi;
    const unsigned short* Bg = p.Bm + zl * p.sBzLo + zh * p.sBzHi;

    // Per-thread staging slots: 4 chunks of 8 bf16 for A and for B (both K-major).
    // Global pointers are advanced by GBK per tile (strength-reduced addressing).
    const unsigned short* gA[4];
    const unsigned short* gB[4];
    uint32_t lA[4], lB[4];
    {
        const uint32_t asBase = (uint32_t)(uintptr_t)(void*)As;
        const uint32_t bsBase = (uint32_t)(uintptr_t)(void*)Bs;
#pragma unroll
        for (int i = 0; i < 4; ++i) {
            const int c   = tid + i * 256;       // 1024 chunks
            const int row = c >> 3;              // 8 chunks per 64-wide row
            const int kc  = (c & 7) << 3;
            gA[i] = Ag + (m0 + row) * p.lda + kc;
            gB[i] = Bg + (n0 + row) * p.ldb + kc;
            lA[i] = asBase + (uint32_t)(row * LDSS + kc) * 2u;
            lB[i] = bsBase + (uint32_t)(row * LDSS + kc) * 2u;
        }
    }

    v8f acc[2][4];
#pragma unroll
    for (int i = 0; i < 2; ++i)
#pragma unroll
        for (int j = 0; j < 4; ++j)
            acc[i][j] = (v8f){0.f,0.f,0.f,0.f,0.f,0.f,0.f,0.f};

    const int nK = p.K >> 6;

    auto stage_adv = [&](int buf) {          // issue async DMA, advance pointers
        const uint32_t boff = (uint32_t)buf * (TILE_USH * 2u);
#pragma unroll
        for (int i = 0; i < 4; ++i) { async_copy16(lA[i] + boff, gA[i]); gA[i] += GBK; }
#pragma unroll
        for (int i = 0; i < 4; ++i) { async_copy16(lB[i] + boff, gB[i]); gB[i] += GBK; }
    };
    auto compute = [&](int buf) {
        const int bo = buf * TILE_USH;
#pragma unroll
        for (int s = 0; s < GBK; s += 32) {
            v16bf a[2], b[4];
#pragma unroll
            for (int i = 0; i < 2; ++i)
                a[i] = load_fragA(&As[bo + (wm + i * 16 + (lane & 15)) * LDSS], s, lane);
#pragma unroll
            for (int j = 0; j < 4; ++j)
                b[j] = load_fragB(&Bs[bo + (wn + j * 16 + (lane & 15)) * LDSS], s, lane);
#pragma unroll
            for (int i = 0; i < 2; ++i)
#pragma unroll
                for (int j = 0; j < 4; ++j)
                    acc[i][j] = __builtin_amdgcn_wmma_f32_16x16x32_bf16(
                        false, a[i], false, b[j], (short)0, acc[i][j], false, false);
        }
    };

    // ---- 2-stage software pipeline (peeled: steady-state loop is branch-free)
    stage_adv(0);
    wait_async0();
    __syncthreads();
    for (int kt = 0; kt < nK - 1; ++kt) {
        stage_adv((kt + 1) & 1);             // DMA for next tile overlaps WMMA
        compute(kt & 1);
        wait_async0();
        __syncthreads();
    }
    compute((nK - 1) & 1);

    // ---- epilogue: C/D layout -> VGPR r = row r (+8 upper half-wave), N = lane%16
    const long long cbase = zl * p.sCzLo + zh * p.sCzHi;
    const int nlo = lane & 15;
    const int rhi = (lane >> 4) * 8;
#pragma unroll
    for (int i = 0; i < 2; ++i) {
#pragma unroll
        for (int j = 0; j < 4; ++j) {
            const long long col = n0 + wn + j * 16 + nlo;
            const float bia = p.bias ? p.bias[col] : 0.f;
            if (p.storeCT) {
                // transposed bf16 store: 8 consecutive rows at fixed col -> one 16B store
                const long long rowBase = m0 + wm + i * 16 + rhi;
                u16x8 t;
#pragma unroll
                for (int r = 0; r < 8; ++r) t[r] = f32_to_bf16(acc[i][j][r] + bia);
                *reinterpret_cast<u16x8*>(&p.Ch[cbase + col * p.ldct + rowBase]) = t;
            } else {
#pragma unroll
                for (int r = 0; r < 8; ++r) {
                    const long long row = m0 + wm + i * 16 + rhi + r;
                    float v = acc[i][j][r] + bia;
                    if (p.resid) v += p.resid[row * p.ldr + col];
                    if (p.relu)  v = fmaxf(v, 0.f);
                    const long long o = cbase + row * p.ldc + col;
                    if (p.Cf) p.Cf[o] = v;
                    if (p.Ch) p.Ch[o] = f32_to_bf16(v);
                }
            }
        }
    }
}

// ---------------------------------------------------------------------------
// Causal masked softmax: P = softmax(S * D^-0.5, causal), bf16 out.
// grid = (T, BH), block = 256; row cached in registers (2048 = 256*8).
// ---------------------------------------------------------------------------
__global__ __launch_bounds__(256) void softmax_kernel(const float* __restrict__ S,
                                                      unsigned short* __restrict__ P) {
    const int m = blockIdx.x;        // query position
    const int z = blockIdx.y;        // h*B + b
    const float scale = 0.03125f;    // (n_embd=1024)^-0.5 per reference
    const float* row = S + ((long long)z * cT + m) * cT;
    unsigned short* prow = P + ((long long)z * cT + m) * cT;
    const int tid = threadIdx.x;
    __shared__ float red[8];

    float rv[8];
    float lmax = -3.4e38f;
#pragma unroll
    for (int i = 0; i < 8; ++i) {
        const int s = tid + i * 256;
        rv[i] = (s <= m) ? row[s] * scale : -3.4e38f;
        lmax = fmaxf(lmax, rv[i]);
    }
    for (int o = 16; o; o >>= 1) lmax = fmaxf(lmax, __shfl_xor(lmax, o));
    if ((tid & 31) == 0) red[tid >> 5] = lmax;
    __syncthreads();
    float gmax = red[0];
#pragma unroll
    for (int w = 1; w < 8; ++w) gmax = fmaxf(gmax, red[w]);
    __syncthreads();

    float lsum = 0.f;
#pragma unroll
    for (int i = 0; i < 8; ++i) {
        rv[i] = __expf(rv[i] - gmax);    // masked lanes: exp(-huge) == 0
        lsum += rv[i];
    }
    for (int o = 16; o; o >>= 1) lsum += __shfl_xor(lsum, o);
    if ((tid & 31) == 0) red[tid >> 5] = lsum;
    __syncthreads();
    float gsum = 0.f;
#pragma unroll
    for (int w = 0; w < 8; ++w) gsum += red[w];
    const float inv = 1.f / gsum;

#pragma unroll
    for (int i = 0; i < 8; ++i)
        prow[tid + i * 256] = f32_to_bf16(rv[i] * inv);
}

// ---------------------------------------------------------------------------
// Host orchestration
// ---------------------------------------------------------------------------
extern "C" void kernel_launch(void* const* d_in, const int* in_sizes, int n_in,
                              void* d_out, int out_size, void* d_ws, size_t ws_size,
                              hipStream_t stream) {
    (void)in_sizes; (void)n_in; (void)out_size; (void)ws_size;

    const int*   idx   = (const int*)  d_in[0];
    const float* tok   = (const float*)d_in[1];
    const float* pos   = (const float*)d_in[2];
    const float* Wq    = (const float*)d_in[3];
    const float* Wk    = (const float*)d_in[4];
    const float* Wv    = (const float*)d_in[5];
    const float* Wproj = (const float*)d_in[6];
    const float* bproj = (const float*)d_in[7];
    const float* W1    = (const float*)d_in[8];
    const float* b1    = (const float*)d_in[9];
    const float* W2    = (const float*)d_in[10];
    const float* b2    = (const float*)d_in[11];
    const float* Wlm   = (const float*)d_in[12];
    float* out = (float*)d_out;

    // ---- workspace layout (bytes); total ~435 MB ----
    char* ws = (char*)d_ws;
    long long off = 0;
    auto take = [&](long long bytes) { char* p = ws + off; off += (bytes + 255) & ~255LL; return p; };
    float*          xf   = (float*)         take(cBT * cD * 4);            // fp32 activations
    unsigned short* xh   = (unsigned short*)take(cBT * cD * 2);            // bf16 activations
    unsigned short* qb   = (unsigned short*)take(cH * cBT * cDH * 2);      // [H,B,T,DH]
    unsigned short* kb   = (unsigned short*)take(cH * cBT * cDH * 2);      // [H,B,T,DH]
    unsigned short* vtb  = (unsigned short*)take(cH * cDH * cBT * 2);      // V^T: [H,DH,B*T]
    unsigned short* ob   = (unsigned short*)take(cBT * cD * 2);            // concat heads [B,T,D]
    unsigned short* h1   = (unsigned short*)take(cBT * cFF * 2);
    float*          sc   = (float*)         take(cBH * cT * cT * 4);       // scores
    unsigned short* pr   = (unsigned short*)take(cBH * cT * cT * 2);       // probs bf16
    unsigned short* wq_t = (unsigned short*)take(cL * cH * cDH * cD * 2);  // [l,h][DH,D]
    unsigned short* wk_t = (unsigned short*)take(cL * cH * cDH * cD * 2);
    unsigned short* wv_t = (unsigned short*)take(cL * cH * cDH * cD * 2);
    unsigned short* wp_t = (unsigned short*)take(cL * cD * cD * 2);        // [l][D,D] (N-major)
    unsigned short* w1_t = (unsigned short*)take(cL * cFF * cD * 2);       // [l][FF,D]
    unsigned short* w2_t = (unsigned short*)take(cL * cD * cFF * 2);       // [l][D,FF]
    unsigned short* wl_t = (unsigned short*)take(cV * cD * 2);             // [V,D]

    auto cvtT = [&](const float* s, unsigned short* d, int K, int N, int nmat) {
        dim3 grid((unsigned)(N / 32), (unsigned)(K / 32), (unsigned)nmat);
        cvtT_f32_bf16<<<grid, dim3(32, 8), 0, stream>>>(s, d, K, N);
    };
    cvtT(Wq,    wq_t, (int)cD,  (int)cDH, (int)(cL * cH));
    cvtT(Wk,    wk_t, (int)cD,  (int)cDH, (int)(cL * cH));
    cvtT(Wv,    wv_t, (int)cD,  (int)cDH, (int)(cL * cH));
    cvtT(Wproj, wp_t, (int)cD,  (int)cD,  (int)cL);
    cvtT(W1,    w1_t, (int)cD,  (int)cFF, (int)cL);
    cvtT(W2,    w2_t, (int)cFF, (int)cD,  (int)cL);
    cvtT(Wlm,   wl_t, (int)cD,  (int)cV,  1);

    embed_kernel<<<(unsigned)((cBT * cD) / 256), 256, 0, stream>>>(idx, tok, pos, xf, xh);

    auto gemm = [&](GemmParams p, int gz) {
        dim3 grid((unsigned)(p.N / GBN), (unsigned)(p.M / GBM), (unsigned)gz);
        gemm_bf16_kernel<<<grid, 256, 0, stream>>>(p);
    };
    auto mk = [&]() {
        GemmParams p{};
        p.zmod = 1;
        return p;
    };

    for (int l = 0; l < (int)cL; ++l) {
        // ---- Q,K projections: z = head; out [H,B*T,DH] ----
        const unsigned short* wsrc[2] = {wq_t, wk_t};
        unsigned short*       dst[2]  = {qb, kb};
        for (int s = 0; s < 2; ++s) {
            GemmParams p = mk();
            p.A = xh;  p.lda = cD;
            p.Bm = wsrc[s] + (long long)l * cH * cDH * cD; p.ldb = cD; p.sBzLo = cDH * cD;
            p.Ch = dst[s]; p.ldc = cDH;
            p.zmod = (int)cH; p.sCzLo = cBT * cDH;
            p.M = (int)cBT; p.N = (int)cDH; p.K = (int)cD;
            gemm(p, (int)cH);
        }
        // ---- V projection, stored transposed: V^T [H, DH, B*T] ----
        {
            GemmParams p = mk();
            p.A = xh;  p.lda = cD;
            p.Bm = wv_t + (long long)l * cH * cDH * cD; p.ldb = cD; p.sBzLo = cDH * cD;
            p.Ch = vtb; p.storeCT = 1; p.ldct = cBT;
            p.zmod = (int)cH; p.sCzLo = cDH * cBT;
            p.M = (int)cBT; p.N = (int)cDH; p.K = (int)cD;
            gemm(p, (int)cH);
        }
        // ---- scores = Q K^T, z = h*B + b (K rows are K-major already) ----
        {
            GemmParams p = mk();
            p.A = qb;  p.lda = cDH; p.sAzLo = cT * cDH;
            p.Bm = kb; p.ldb = cDH; p.sBzLo = cT * cDH;
            p.Cf = sc; p.ldc = cT;
            p.zmod = (int)cBH; p.sCzLo = cT * cT;
            p.M = (int)cT; p.N = (int)cT; p.K = (int)cDH;
            gemm(p, (int)cBH);
        }
        // ---- masked softmax -> bf16 probs ----
        softmax_kernel<<<dim3((unsigned)cT, (unsigned)cBH), 256, 0, stream>>>(sc, pr);
        // ---- O = P V : B = V^T slice [DH, T]; heads scattered into [B,T,H*DH] ----
        {
            GemmParams p = mk();
            p.A = pr;   p.lda = cT;  p.sAzLo = cT * cT;  p.sAzHi = cB * cT * cT;
            p.Bm = vtb; p.ldb = cBT; p.sBzLo = cT;       p.sBzHi = cDH * cBT;
            p.Ch = ob;  p.ldc = cD;
            p.zmod = (int)cB; p.sCzLo = cT * cD; p.sCzHi = cDH;   // base = b*T*D + h*DH
            p.M = (int)cT; p.N = (int)cDH; p.K = (int)cT;
            gemm(p, (int)cBH);
        }
        // ---- x = x + O @ Wproj + bproj ----
        {
            GemmParams p = mk();
            p.A = ob; p.lda = cD;
            p.Bm = wp_t + (long long)l * cD * cD; p.ldb = cD;
            p.bias = bproj + (long long)l * cD;
            p.resid = xf; p.ldr = cD;
            p.Cf = xf; p.Ch = xh; p.ldc = cD;
            p.M = (int)cBT; p.N = (int)cD; p.K = (int)cD;
            gemm(p, 1);
        }
        // ---- h1 = relu(x @ W1 + b1) ----
        {
            GemmParams p = mk();
            p.A = xh; p.lda = cD;
            p.Bm = w1_t + (long long)l * cFF * cD; p.ldb = cD;
            p.bias = b1 + (long long)l * cFF; p.relu = 1;
            p.Ch = h1; p.ldc = cFF;
            p.M = (int)cBT; p.N = (int)cFF; p.K = (int)cD;
            gemm(p, 1);
        }
        // ---- x = x + h1 @ W2 + b2 ----
        {
            GemmParams p = mk();
            p.A = h1; p.lda = cFF;
            p.Bm = w2_t + (long long)l * cD * cFF; p.ldb = cFF;
            p.bias = b2 + (long long)l * cD;
            p.resid = xf; p.ldr = cD;
            p.Cf = xf; p.Ch = xh; p.ldc = cD;
            p.M = (int)cBT; p.N = (int)cD; p.K = (int)cFF;
            gemm(p, 1);
        }
    }
    // ---- logits = x @ Wlm  (dominant GEMM: 268 GFLOP) ----
    {
        GemmParams p = mk();
        p.A = xh; p.lda = cD;
        p.Bm = wl_t; p.ldb = cD;
        p.Cf = out; p.ldc = cV;
        p.M = (int)cBT; p.N = (int)cV; p.K = (int)cD;
        gemm(p, 1);
    }
}
